// ColumnarTransformerBlock_31988916421132
// MI455X (gfx1250) — compile-verified, matrix operand order checked by reading, not run on hardware
//
#include <hip/hip_runtime.h>
#include <hip/hip_bf16.h>

// ---------------------------------------------------------------------------
// ColumnarTransformerBlock for MI455X (gfx1250, wave32, WMMA).
// All GEMMs (QKV / Wo / Wgu / Wd and attention S=QK^T, O=PV) run on
// v_wmma_f32_16x16x32_f16 with f32 accumulation; f32 operands are converted
// to f16 while staging into LDS / fragments.
// GEMM: 64x128 block tile, K-chunk 64, 8 waves x (32x32) -> 8 WMMAs per wave
// per barrier pair (high matrix-op density, the MI455X bottleneck resource).
// ---------------------------------------------------------------------------

#define B_N   4
#define SEQ   1024
#define DMODEL 1024
#define NHEAD 16
#define HD    64
#define SCOLS 8
#define TOPK  2
#define INTER 768
#define EPSV  1e-5f

typedef __attribute__((ext_vector_type(16))) _Float16 v16h;
typedef __attribute__((ext_vector_type(8)))  _Float16 v8h;
typedef __attribute__((ext_vector_type(8)))  float    v8f;

static __device__ __forceinline__ v8f wmma_f16(v16h a, v16h b, v8f c) {
  // (neg_a, A, neg_b, B, c_mod, C, reuse_a, reuse_b)
  return __builtin_amdgcn_wmma_f32_16x16x32_f16(false, a, false, b,
                                                (short)0, c, false, false);
}

// ---------------------------------------------------------------------------
// 1) mean over sequence:  mean[b][d] = avg_s hidden[b][s][d]
// ---------------------------------------------------------------------------
__global__ __launch_bounds__(256) void mean_kernel(const float* __restrict__ hidden,
                                                   float* __restrict__ meanb) {
  int i = blockIdx.x * 256 + threadIdx.x;          // 0 .. B*D-1
  int b = i >> 10, d = i & (DMODEL - 1);
  const float* p = hidden + (long)b * SEQ * DMODEL + d;
  float s = 0.f;
  for (int t = 0; t < SEQ; ++t) s += p[(long)t * DMODEL];
  meanb[i] = s * (1.0f / SEQ);
}

// ---------------------------------------------------------------------------
// 2) routing: logits = mean @ Wr^T / clamp(temp); top-2 + softmax weights
// ---------------------------------------------------------------------------
__global__ __launch_bounds__(32) void routing_kernel(const float* __restrict__ meanb,
                                                     const float* __restrict__ Wr,
                                                     const float* __restrict__ temperature,
                                                     float* __restrict__ logits_out,
                                                     float* __restrict__ flat_w,
                                                     int* __restrict__ flat_idx) {
  __shared__ float lg[B_N * SCOLS];
  const int lane = threadIdx.x;                    // 32 lanes, one (b,c) each
  const int b = lane >> 3, c = lane & 7;
  const float temp = fminf(fmaxf(temperature[0], 0.1f), 10.0f);
  const float* m = meanb + b * DMODEL;
  const float* w = Wr + c * DMODEL;
  float dot = 0.f;
  for (int d = 0; d < DMODEL; ++d) dot += m[d] * w[d];
  const float logit = dot / temp;
  logits_out[b * SCOLS + c] = logit;
  lg[lane] = logit;
  __syncthreads();
  if (lane < B_N) {
    float v1 = -1e30f, v2 = -1e30f;
    int i1 = 0, i2 = 0;
    for (int cc = 0; cc < SCOLS; ++cc) {
      float v = lg[lane * SCOLS + cc];
      if (v > v1)      { v2 = v1; i2 = i1; v1 = v; i1 = cc; }
      else if (v > v2) { v2 = v;  i2 = cc; }
    }
    const float e = __expf(v2 - v1);
    flat_idx[2 * lane + 0] = i1;
    flat_idx[2 * lane + 1] = i2;
    flat_w[2 * lane + 0] = 1.0f / (1.0f + e);
    flat_w[2 * lane + 1] = e / (1.0f + e);
  }
}

// ---------------------------------------------------------------------------
// 3) Generic expert GEMM:  Y[n] = X[n/x_div] @ W[eidx[n]]
//    Block tile 64(M) x 128(N), K-chunk 64.  8 waves, each owns 32x32 out
//    (2x2 WMMA tiles x 2 K-halves = 8 WMMAs per chunk).
//    WMMA fragment layout (16-bit A/B, wave32):
//      lane L: row/col = L&15, kb = (L>>4)*8,
//      elements j<8 -> K = kb+j,  j>=8 -> K = 16+kb+(j-8)
// ---------------------------------------------------------------------------
#define GEMM_TM 64
#define GEMM_TN 128
#define GEMM_KC 64
#define LDA_H   72   // f16 row stride (144 bytes, 16B multiple)
#define LDB_H   72

__global__ __launch_bounds__(256) void gemm_expert_kernel(
    const float* __restrict__ X, int x_div, long x_stride,
    const float* __restrict__ W, long w_stride,
    const int* __restrict__ eidx,
    float* __restrict__ Y, long y_stride,
    int K, int N) {
  __shared__ __align__(16) _Float16 sA[GEMM_TM * LDA_H];   // [m][k]
  __shared__ __align__(16) _Float16 sB[GEMM_TN * LDB_H];   // transposed: [n][k]

  const int n = blockIdx.z;
  const float* Xb = X + (long)(n / x_div) * x_stride;
  const float* Wb = W + (long)eidx[n] * w_stride;
  float* Yb = Y + (long)n * y_stride;

  const int tid = threadIdx.x;
  const int wid = tid >> 5, lane = tid & 31;
  const int tm = (wid >> 2) * 32;        // wave M sub-tile: 0 / 32
  const int tn = (wid & 3) * 32;         // wave N sub-tile: 0,32,64,96
  const int gm0 = blockIdx.y * GEMM_TM;
  const int gn0 = blockIdx.x * GEMM_TN;
  const int l15 = lane & 15;
  const int kb = (lane >> 4) * 8;

  v8f acc[2][2] = {};

  for (int k0 = 0; k0 < K; k0 += GEMM_KC) {
    // ---- stage A tile (64 rows x 64 K) as f16, row-major: 4 float4/thread
#pragma unroll
    for (int i = 0; i < 4; ++i) {
      const int idx = tid + i * 256;         // 0..1023
      const int r = idx >> 4;                // 64 rows
      const int c4 = (idx & 15) * 4;         // 16 float4 per row
      const float4 v = *(const float4*)(Xb + (long)(gm0 + r) * K + k0 + c4);
      _Float16* dst = &sA[r * LDA_H + c4];
      dst[0] = (_Float16)v.x; dst[1] = (_Float16)v.y;
      dst[2] = (_Float16)v.z; dst[3] = (_Float16)v.w;
    }
    // ---- stage B tile (64 K x 128 N) as f16, TRANSPOSED [n][k]: 8 float4/thread
#pragma unroll
    for (int i = 0; i < 8; ++i) {
      const int idx = tid + i * 256;         // 0..2047
      const int kr = idx >> 5;               // 64 K rows
      const int c4 = (idx & 31) * 4;         // 32 float4 per row
      const float4 v = *(const float4*)(Wb + (long)(k0 + kr) * N + gn0 + c4);
      sB[(c4 + 0) * LDB_H + kr] = (_Float16)v.x;
      sB[(c4 + 1) * LDB_H + kr] = (_Float16)v.y;
      sB[(c4 + 2) * LDB_H + kr] = (_Float16)v.z;
      sB[(c4 + 3) * LDB_H + kr] = (_Float16)v.w;
    }
    // L2 prefetch of next K chunk
    if (k0 + GEMM_KC < K) {
      __builtin_prefetch(Xb + (long)(gm0 + (tid >> 2)) * K + k0 + GEMM_KC, 0, 1);
      __builtin_prefetch(Wb + (long)(k0 + GEMM_KC + (tid >> 2)) * N + gn0, 0, 1);
    }
    __syncthreads();

    // ---- fragments: 4 A (2 m-tiles x 2 k-halves), 4 B; 8 WMMAs ----
    v16h af[2][2], bf[2][2];
#pragma unroll
    for (int mi = 0; mi < 2; ++mi) {
#pragma unroll
      for (int kh = 0; kh < 2; ++kh) {
        const _Float16* base = &sA[(tm + mi * 16 + l15) * LDA_H + kh * 32 + kb];
        const v8h lo = *(const v8h*)base;
        const v8h hi = *(const v8h*)(base + 16);
#pragma unroll
        for (int j = 0; j < 8; ++j) { af[mi][kh][j] = lo[j]; af[mi][kh][j + 8] = hi[j]; }
      }
    }
#pragma unroll
    for (int ni = 0; ni < 2; ++ni) {
#pragma unroll
      for (int kh = 0; kh < 2; ++kh) {
        const _Float16* base = &sB[(tn + ni * 16 + l15) * LDB_H + kh * 32 + kb];
        const v8h lo = *(const v8h*)base;
        const v8h hi = *(const v8h*)(base + 16);
#pragma unroll
        for (int j = 0; j < 8; ++j) { bf[ni][kh][j] = lo[j]; bf[ni][kh][j + 8] = hi[j]; }
      }
    }
#pragma unroll
    for (int mi = 0; mi < 2; ++mi)
#pragma unroll
      for (int ni = 0; ni < 2; ++ni) {
        acc[mi][ni] = wmma_f16(af[mi][0], bf[ni][0], acc[mi][ni]);
        acc[mi][ni] = wmma_f16(af[mi][1], bf[ni][1], acc[mi][ni]);
      }
    __syncthreads();
  }

  // ---- write D: element r -> row r + 8*(lane>>4), col = lane&15 ----
#pragma unroll
  for (int mi = 0; mi < 2; ++mi) {
    const int row0 = gm0 + tm + mi * 16 + (lane >> 4) * 8;
#pragma unroll
    for (int ni = 0; ni < 2; ++ni) {
      const int col = gn0 + tn + ni * 16 + l15;
#pragma unroll
      for (int r = 0; r < 8; ++r)
        Yb[(long)(row0 + r) * N + col] = acc[mi][ni][r];
    }
  }
}

// ---------------------------------------------------------------------------
// 4) RoPE applied in-place to q and k halves of the qkv buffer
// ---------------------------------------------------------------------------
__global__ __launch_bounds__(256) void rope_kernel(float* __restrict__ qkv,
                                                   const float* __restrict__ cosb,
                                                   const float* __restrict__ sinb) {
  const int n = blockIdx.y, s = blockIdx.x;
  float* row = qkv + (long)n * SEQ * (3 * DMODEL) + (long)s * (3 * DMODEL);
#pragma unroll
  for (int i = 0; i < 4; ++i) {
    const int tsk = threadIdx.x + i * 256;   // 1024 pair-tasks: q then k
    const int which = tsk >> 9;              // 0 = q, 1 = k
    const int rem = tsk & 511;
    const int h = rem >> 5, p = rem & 31;    // head, pair index
    float* ptr = row + which * DMODEL + h * HD;
    const float c0 = cosb[s * HD + p],       s0 = sinb[s * HD + p];
    const float c1 = cosb[s * HD + 32 + p],  s1 = sinb[s * HD + 32 + p];
    const float x1 = ptr[p], x2 = ptr[32 + p];
    ptr[p]      = x1 * c0 - x2 * s0;
    ptr[32 + p] = x2 * c1 + x1 * s1;
  }
}

// ---------------------------------------------------------------------------
// 5) Flash attention: one wave per 16-query tile of one (n, head).
//    S = (Q/8) K^T via 2 chained WMMAs (K=64); online softmax with
//    cross-lane reductions; P re-laid-out through LDS (s_wait_dscnt);
//    O += P V via 4 WMMAs (K padded 16->32 with zeros).
// ---------------------------------------------------------------------------
__global__ __launch_bounds__(128) void attn_kernel(const float* __restrict__ qkv,
                                                   float* __restrict__ ctx) {
  const int n = blockIdx.z, h = blockIdx.y;
  const int wid = threadIdx.x >> 5, lane = threadIdx.x & 31;
  const int q0 = (blockIdx.x * 4 + wid) * 16;
  const int l15 = lane & 15, hl = lane >> 4, kb = hl * 8;

  const long base = (long)n * SEQ * (3 * DMODEL);
  const float* qp = qkv + base + h * HD;
  const float* kp = qkv + base + DMODEL + h * HD;
  const float* vp = qkv + base + 2 * DMODEL + h * HD;

  __shared__ __align__(16) _Float16 sP[4][16][24];   // 24 halfs = 48B rows

  // Q fragments (pre-scaled by 1/sqrt(HD))
  v16h qa0, qa1;
  {
    const float scale = 0.125f;
    const float* qrow = qp + (long)(q0 + l15) * (3 * DMODEL);
#pragma unroll
    for (int j = 0; j < 8; ++j) {
      qa0[j]     = (_Float16)(qrow[kb + j] * scale);
      qa0[j + 8] = (_Float16)(qrow[16 + kb + j] * scale);
      qa1[j]     = (_Float16)(qrow[32 + kb + j] * scale);
      qa1[j + 8] = (_Float16)(qrow[48 + kb + j] * scale);
    }
  }

  float mrow[8], lrow[8];
  v8f acc0 = {}, acc1 = {}, acc2 = {}, acc3 = {};
#pragma unroll
  for (int r = 0; r < 8; ++r) { mrow[r] = -1e30f; lrow[r] = 0.f; }

  for (int kc = 0; kc < SEQ / 16; ++kc) {
    // K^T fragment: col = key, K dim = head dim
    v16h kf0, kf1;
    {
      const float* krow = kp + (long)(kc * 16 + l15) * (3 * DMODEL);
#pragma unroll
      for (int j = 0; j < 8; ++j) {
        kf0[j]     = (_Float16)krow[kb + j];
        kf0[j + 8] = (_Float16)krow[16 + kb + j];
        kf1[j]     = (_Float16)krow[32 + kb + j];
        kf1[j + 8] = (_Float16)krow[48 + kb + j];
      }
    }
    v8f s = {};
    s = wmma_f16(qa0, kf0, s);
    s = wmma_f16(qa1, kf1, s);

    // online softmax, rows live in 16-lane halves
    float p[8];
#pragma unroll
    for (int r = 0; r < 8; ++r) {
      float v = s[r];
      for (int off = 1; off < 16; off <<= 1)
        v = fmaxf(v, __shfl_xor(v, off, 32));
      const float mnew = fmaxf(mrow[r], v);
      const float alpha = __expf(mrow[r] - mnew);
      const float pe = __expf(s[r] - mnew);
      float rs = pe;
      for (int off = 1; off < 16; off <<= 1)
        rs += __shfl_xor(rs, off, 32);
      lrow[r] = lrow[r] * alpha + rs;
      mrow[r] = mnew;
      p[r] = pe;
      acc0[r] *= alpha; acc1[r] *= alpha; acc2[r] *= alpha; acc3[r] *= alpha;
    }

    // P (D-layout) -> LDS -> A-layout fragment, zero-padded K 16..31
#pragma unroll
    for (int r = 0; r < 8; ++r)
      sP[wid][r + hl * 8][l15] = (_Float16)p[r];
    __asm__ volatile("s_wait_dscnt 0" ::: "memory");   // wave-local LDS RAW

    v16h pa;
    {
      const v8h lo = *(const v8h*)&sP[wid][l15][kb];
#pragma unroll
      for (int j = 0; j < 8; ++j) { pa[j] = lo[j]; pa[j + 8] = (_Float16)0.f; }
    }

    // O += P @ V, 4 N-sub-tiles of the 64-wide head dim
#pragma unroll
    for (int t = 0; t < 4; ++t) {
      v16h vb;
#pragma unroll
      for (int j = 0; j < 8; ++j) {
        vb[j] = (_Float16)vp[(long)(kc * 16 + kb + j) * (3 * DMODEL) + t * 16 + l15];
        vb[j + 8] = (_Float16)0.f;
      }
      if (t == 0) acc0 = wmma_f16(pa, vb, acc0);
      else if (t == 1) acc1 = wmma_f16(pa, vb, acc1);
      else if (t == 2) acc2 = wmma_f16(pa, vb, acc2);
      else acc3 = wmma_f16(pa, vb, acc3);
    }
  }

  // finalize: divide by l, scatter to ctx[n][s][h*64+d]
  float* cp = ctx + (long)n * SEQ * DMODEL;
#pragma unroll
  for (int r = 0; r < 8; ++r) {
    const float inv = 1.0f / lrow[r];
    const int row = q0 + r + hl * 8;
    float* crow = cp + (long)row * DMODEL + h * HD;
    crow[0 * 16 + l15] = acc0[r] * inv;
    crow[1 * 16 + l15] = acc1[r] * inv;
    crow[2 * 16 + l15] = acc2[r] * inv;
    crow[3 * 16 + l15] = acc3[r] * inv;
  }
}

// ---------------------------------------------------------------------------
// 6) residual + RMSNorm, writes in-place over Xb:  Xb = rmsnorm(Xa + Xb)
// ---------------------------------------------------------------------------
__global__ __launch_bounds__(256) void rms_residual_kernel(
    const float* __restrict__ Xa, int a_div, long a_stride,
    float* __restrict__ Xb, long b_stride) {
  const int n = blockIdx.y, s = blockIdx.x;
  const float* xa = Xa + (long)(n / a_div) * a_stride + (long)s * DMODEL;
  float* xb = Xb + (long)n * b_stride + (long)s * DMODEL;

  float t[4];
  float ss = 0.f;
#pragma unroll
  for (int i = 0; i < 4; ++i) {
    const int c = threadIdx.x + i * 256;
    t[i] = xa[c] + xb[c];
    ss += t[i] * t[i];
  }
  __shared__ float red[8];
  const int wid = threadIdx.x >> 5, lane = threadIdx.x & 31;
  for (int off = 16; off; off >>= 1) ss += __shfl_xor(ss, off, 32);
  if (lane == 0) red[wid] = ss;
  __syncthreads();
  if (threadIdx.x == 0) {
    float a = 0.f;
    for (int i = 0; i < 8; ++i) a += red[i];
    red[0] = rsqrtf(a * (1.0f / DMODEL) + EPSV);
  }
  __syncthreads();
  const float rms = red[0];
#pragma unroll
  for (int i = 0; i < 4; ++i) {
    const int c = threadIdx.x + i * 256;
    xb[c] = t[i] * rms;
  }
}

// ---------------------------------------------------------------------------
// 7) h = silu(gate) * up    (gu last axis split at INTER)
// ---------------------------------------------------------------------------
__global__ __launch_bounds__(256) void act_kernel(const float* __restrict__ gu,
                                                  float* __restrict__ hact) {
  const long i = (long)blockIdx.x * 256 + threadIdx.x;    // 8*SEQ*INTER
  const long ns = i / INTER;
  const int c = (int)(i % INTER);
  const float* g = gu + ns * (2 * INTER);
  const float gv = g[c], uv = g[INTER + c];
  const float sig = 1.0f / (1.0f + __expf(-gv));
  hact[i] = gv * sig * uv;
}

// ---------------------------------------------------------------------------
// 8) combine experts:  out[b] = w[2b]*x2[2b] + w[2b+1]*x2[2b+1]
// ---------------------------------------------------------------------------
__global__ __launch_bounds__(256) void combine_kernel(const float* __restrict__ x2,
                                                      const float* __restrict__ w,
                                                      float* __restrict__ out) {
  const long i = (long)blockIdx.x * 256 + threadIdx.x;    // B*SEQ*D
  const long per = (long)SEQ * DMODEL;
  const long b = i / per;
  const long rem = i - b * per;
  out[i] = w[2 * b] * x2[(2 * b) * per + rem] +
           w[2 * b + 1] * x2[(2 * b + 1) * per + rem];
}

// ---------------------------------------------------------------------------
// launch
// ---------------------------------------------------------------------------
extern "C" void kernel_launch(void* const* d_in, const int* in_sizes, int n_in,
                              void* d_out, int out_size, void* d_ws, size_t ws_size,
                              hipStream_t stream) {
  (void)in_sizes; (void)n_in; (void)out_size; (void)ws_size;

  const float* hidden = (const float*)d_in[0];   // (4,1024,1024)
  const float* cosb   = (const float*)d_in[1];   // (1024,64)
  const float* sinb   = (const float*)d_in[2];   // (1024,64)
  const float* Wr     = (const float*)d_in[3];   // (8,1024)
  const float* temp   = (const float*)d_in[4];   // (1,)
  const float* Wqkv   = (const float*)d_in[5];   // (8,1024,3072)
  const float* Wo     = (const float*)d_in[6];   // (8,1024,1024)
  const float* Wgu    = (const float*)d_in[7];   // (8,1024,1536)
  const float* Wd     = (const float*)d_in[8];   // (8,768,1024)

  float* out    = (float*)d_out;                            // result (4,1024,1024)
  float* logits = (float*)d_out + (long)B_N * SEQ * DMODEL; // + (4,8)

  // workspace layout (floats); regions reused by lifetime
  float* ws = (float*)d_ws;
  const long WS_R0 = 8192;                                  // qkv region
  const long WS_R1 = WS_R0 + (long)8 * SEQ * (3 * DMODEL);  // ctx region
  const long WS_R2 = WS_R1 + (long)8 * SEQ * DMODEL;        // mlp/x2 region
  float* meanb  = ws;                             // 4096
  float* flat_w = ws + 4096;                      // 8
  int*   flat_i = (int*)(ws + 4104);              // 8
  float* qkv    = ws + WS_R0;                     // (8,1024,3072)   live K1..K3
  float* x1buf  = ws + WS_R0;                     // (8,1024,1024)   attn_out -> x1 (alias)
  float* gubuf  = ws + WS_R0 + (long)8 * SEQ * DMODEL;  // (8,1024,1536)
  float* ctxbuf = ws + WS_R1;                     // (8,1024,1024)   live K3..K4
  float* hbuf   = ws + WS_R1;                     // (8,1024,768)    alias after ctx dead
  float* x2buf  = ws + WS_R2;                     // (8,1024,1024)   mlp_out -> x2 in-place

  // 1-2) routing
  mean_kernel<<<(B_N * DMODEL) / 256, 256, 0, stream>>>(hidden, meanb);
  routing_kernel<<<1, 32, 0, stream>>>(meanb, Wr, temp, logits, flat_w, flat_i);

  // 3) qkv = x_disp @ Wqkv[e]        (M=1024, K=1024, N=3072)
  gemm_expert_kernel<<<dim3(3 * DMODEL / GEMM_TN, SEQ / GEMM_TM, 8), 256, 0, stream>>>(
      hidden, /*x_div=*/2, (long)SEQ * DMODEL,
      Wqkv, (long)DMODEL * 3 * DMODEL, flat_i,
      qkv, (long)SEQ * 3 * DMODEL, DMODEL, 3 * DMODEL);

  // 4) RoPE on q,k
  rope_kernel<<<dim3(SEQ, 8), 256, 0, stream>>>(qkv, cosb, sinb);

  // 5) attention -> ctx
  attn_kernel<<<dim3(SEQ / 64, NHEAD, 8), 128, 0, stream>>>(qkv, ctxbuf);

  // 6) attn_out = ctx @ Wo[e]        (M=1024, K=1024, N=1024) -> x1buf (aliases qkv)
  gemm_expert_kernel<<<dim3(DMODEL / GEMM_TN, SEQ / GEMM_TM, 8), 256, 0, stream>>>(
      ctxbuf, 1, (long)SEQ * DMODEL,
      Wo, (long)DMODEL * DMODEL, flat_i,
      x1buf, (long)SEQ * DMODEL, DMODEL, DMODEL);

  // 7) x1 = rmsnorm(x_disp + attn_out), in-place on x1buf
  rms_residual_kernel<<<dim3(SEQ, 8), 256, 0, stream>>>(
      hidden, 2, (long)SEQ * DMODEL, x1buf, (long)SEQ * DMODEL);

  // 8) gu = x1 @ Wgu[e]              (M=1024, K=1024, N=1536)
  gemm_expert_kernel<<<dim3(2 * INTER / GEMM_TN, SEQ / GEMM_TM, 8), 256, 0, stream>>>(
      x1buf, 1, (long)SEQ * DMODEL,
      Wgu, (long)DMODEL * 2 * INTER, flat_i,
      gubuf, (long)SEQ * 2 * INTER, DMODEL, 2 * INTER);

  // 9) h = silu(gate) * up
  act_kernel<<<(int)(((long)8 * SEQ * INTER) / 256), 256, 0, stream>>>(gubuf, hbuf);

  // 10) mlp_out = h @ Wd[e]          (M=1024, K=768, N=1024)
  gemm_expert_kernel<<<dim3(DMODEL / GEMM_TN, SEQ / GEMM_TM, 8), 256, 0, stream>>>(
      hbuf, 1, (long)SEQ * INTER,
      Wd, (long)INTER * DMODEL, flat_i,
      x2buf, (long)SEQ * DMODEL, INTER, DMODEL);

  // 11) x2 = rmsnorm(x1 + mlp_out), in-place on x2buf
  rms_residual_kernel<<<dim3(SEQ, 8), 256, 0, stream>>>(
      x1buf, 1, (long)SEQ * DMODEL, x2buf, (long)SEQ * DMODEL);

  // 12) weighted combine into output
  combine_kernel<<<(int)(((long)B_N * SEQ * DMODEL) / 256), 256, 0, stream>>>(
      x2buf, flat_w, out);
}
